// AttentionNet_26310969655950
// MI455X (gfx1250) — compile-verified, matrix operand order checked by reading, not run on hardware
//
#include <hip/hip_runtime.h>
#include <hip/hip_bf16.h>
#include <cstdint>

typedef __attribute__((ext_vector_type(16))) _Float16 v16h;
typedef __attribute__((ext_vector_type(8)))  float    v8f;

#define CIN    4
#define SLEN   2048
#define NFLT   256
#define HGATE  1024
#define HID    256
#define SPOOL  341
#define SPPAD  352
#define MROWS  (32 * 341)   // 10912, divisible by 16 (682 tiles)
#define NHEAD  8
#define DHEAD  64
#define MHDIM  512

// ---------------------------------------------------------------- helpers ---

__device__ __forceinline__ v16h load16h(const _Float16* p) {
  union { v16h v; uint4 u[2]; } r;
  r.u[0] = *reinterpret_cast<const uint4*>(p);
  r.u[1] = *reinterpret_cast<const uint4*>(p + 8);
  return r.v;
}

__device__ __forceinline__ float sigm(float x) { return 1.f / (1.f + __expf(-x)); }

__device__ __forceinline__ float redmax16(float x) {
  x = fmaxf(x, __shfl_xor(x, 1, 32));
  x = fmaxf(x, __shfl_xor(x, 2, 32));
  x = fmaxf(x, __shfl_xor(x, 4, 32));
  x = fmaxf(x, __shfl_xor(x, 8, 32));
  return x;
}
__device__ __forceinline__ float redsum16(float x) {
  x += __shfl_xor(x, 1, 32);
  x += __shfl_xor(x, 2, 32);
  x += __shfl_xor(x, 4, 32);
  x += __shfl_xor(x, 8, 32);
  return x;
}

// ------------------------------------------------------------ f32 -> f16 ----

__global__ void f32_to_f16_kernel(const float* __restrict__ s,
                                  _Float16* __restrict__ d, int n) {
  int i = blockIdx.x * 256 + threadIdx.x;
  if (i < n) d[i] = (_Float16)s[i];
}

// --------------------------------------------- conv1d + bn + relu + pool ----

__global__ __launch_bounds__(256) void conv_bn_pool_kernel(
    const float* __restrict__ x, const float* __restrict__ w,
    const float* __restrict__ gamma, const float* __restrict__ beta,
    _Float16* __restrict__ x1) {
  const int bs = blockIdx.x;
  const int b = bs / SPOOL, sp = bs % SPOOL;
  __shared__ float xs[CIN][18];
  const int tid = threadIdx.x;
  const int base = sp * 6 - 6;
  if (tid < CIN * 18) {
    int c = tid / 18, o = tid % 18;
    int pos = base + o;
    xs[c][o] = (pos >= 0 && pos < SLEN) ? x[((size_t)b * CIN + c) * SLEN + pos] : 0.f;
  }
  __syncthreads();
  const int f = tid;
  float wr[CIN * 13];
#pragma unroll
  for (int i = 0; i < CIN * 13; ++i) wr[i] = w[(size_t)f * CIN * 13 + i];
  const float scale = gamma[f] * rsqrtf(1.f + 1e-5f);
  const float bb = beta[f];
  float best = -1e30f;
#pragma unroll
  for (int j = 0; j < 6; ++j) {
    float acc = 0.f;
#pragma unroll
    for (int c = 0; c < CIN; ++c)
#pragma unroll
      for (int kk = 0; kk < 13; ++kk)
        acc += xs[c][j + kk] * wr[c * 13 + kk];
    best = fmaxf(best, fmaxf(acc * scale + bb, 0.f));
  }
  x1[(size_t)bs * NFLT + f] = (_Float16)best;
}

// ------------------------------------------------- pooled relative position -

__global__ void rpe_kernel(float* __restrict__ rpe) {
  int i = blockIdx.x * 256 + threadIdx.x;
  if (i >= SPOOL * SPOOL) return;
  int q = i / SPOOL, kk = i % SPOOL;
  int r0 = max(q * 6 - 1, 0), r1 = min(q * 6 + 7, SLEN - 1);
  int c0 = max(kk * 6 - 1, 0), c1 = min(kk * 6 + 7, SLEN - 1);
  int m = max(max(abs(r0 - c0), abs(r0 - c1)), max(abs(r1 - c0), abs(r1 - c1)));
  rpe[i] = (float)m / (float)(SLEN - 1);
}

// ------------------------------------------------------- generic WMMA GEMM --
// out[M,N] = A[M,K](f16) @ W[N,K](f16)^T + bias.
// One wave per 16x64 tile (4 accumulators, A reused), double-buffered K loop.
// OUTMODE: 0 = f32 row-major, 1 = f16 row-major, 2 = f16 V-transposed
//          (out[b][col][s], row stride SPPAD, for the attention PV operand).

template <int OUTMODE, bool RELU>
__global__ __launch_bounds__(32) void gemm_wmma_kernel(
    const _Float16* __restrict__ A, const _Float16* __restrict__ W,
    const float* __restrict__ bias, void* __restrict__ out, int N, int K) {
  const int lane = threadIdx.x & 31;
  const int nn = lane & 15, hg = lane >> 4;
  const int m0 = blockIdx.x << 4, n0 = blockIdx.y << 6;
  const _Float16* arow = A + (size_t)(m0 + nn) * K + (hg << 4);
  const _Float16* wrow = W + (size_t)(n0 + nn) * K + (hg << 4);

  v8f acc[4] = {};
  v16h a = load16h(arow);
  v16h b[4];
#pragma unroll
  for (int tt = 0; tt < 4; ++tt) b[tt] = load16h(wrow + (size_t)tt * 16 * K);

  int k0 = 0;
  for (; k0 + 32 < K; k0 += 32) {
    v16h an = load16h(arow + k0 + 32);
    v16h bn[4];
#pragma unroll
    for (int tt = 0; tt < 4; ++tt)
      bn[tt] = load16h(wrow + (size_t)tt * 16 * K + k0 + 32);
#pragma unroll
    for (int tt = 0; tt < 4; ++tt)
      acc[tt] = __builtin_amdgcn_wmma_f32_16x16x32_f16(false, a, false, b[tt],
                                                       (short)0, acc[tt], false, false);
    a = an;
#pragma unroll
    for (int tt = 0; tt < 4; ++tt) b[tt] = bn[tt];
  }
#pragma unroll
  for (int tt = 0; tt < 4; ++tt)
    acc[tt] = __builtin_amdgcn_wmma_f32_16x16x32_f16(false, a, false, b[tt],
                                                     (short)0, acc[tt], false, false);

  const int rbase = m0 + (hg << 3);
#pragma unroll
  for (int tt = 0; tt < 4; ++tt) {
    const int col = n0 + tt * 16 + nn;
    const float bv = bias ? bias[col] : 0.f;
#pragma unroll
    for (int r = 0; r < 8; ++r) {
      float xv = acc[tt][r] + bv;
      if (RELU) xv = fmaxf(xv, 0.f);
      const int row = rbase + r;
      if (OUTMODE == 0) {
        reinterpret_cast<float*>(out)[(size_t)row * N + col] = xv;
      } else if (OUTMODE == 1) {
        reinterpret_cast<_Float16*>(out)[(size_t)row * N + col] = (_Float16)xv;
      } else {
        int bb = row / SPOOL, s = row % SPOOL;
        reinterpret_cast<_Float16*>(out)[((size_t)bb * 512 + col) * SPPAD + s] =
            (_Float16)xv;
      }
    }
  }
}

// ------------------------------------------------------ fused LSTM cell -----
// One launch per scan step. Each wave: 16 rows x 16 hidden units, all 4 gate
// tiles (32 WMMAs over K=256, fully unrolled + double-buffered), + precomputed
// input projection (bias merged), gate nonlinearities, c/h update. h is
// ping-ponged (hin/hout) to avoid cross-block RAW within a step.
// blockIdx.z = direction; bwd consumes gin row (31-t) and writes slot (31-t).

__global__ __launch_bounds__(32) void lstm_cell_kernel(
    const _Float16* __restrict__ gin_base,   // [2][MROWS][1024]
    const _Float16* __restrict__ whh_base,   // [2][1024][256]
    const _Float16* __restrict__ hin_base,   // [2][352][256]
    _Float16* __restrict__ hout_base,        // [2][352][256]
    float* __restrict__ c_base,              // [2][352][256]
    _Float16* __restrict__ xout,             // [MROWS][512]
    int t_fwd) {
  const int lane = threadIdx.x & 31;
  const int nn = lane & 15, hg = lane >> 4;
  const int dir = blockIdx.z;
  const int t = dir ? (31 - t_fwd) : t_fwd;
  const _Float16* gin = gin_base + (size_t)dir * MROWS * HGATE;
  const _Float16* whh = whh_base + (size_t)dir * HGATE * HID;
  const _Float16* hin = hin_base + (size_t)dir * SPPAD * HID;
  _Float16* hout = hout_base + (size_t)dir * SPPAD * HID;
  float* cst = c_base + (size_t)dir * SPPAD * HID;

  const int m0 = blockIdx.x << 4, n0 = blockIdx.y << 4;
  const _Float16* arow = hin + (size_t)(m0 + nn) * HID + (hg << 4);
  const _Float16* wrow = whh + (size_t)(n0 + nn) * HID + (hg << 4);

  v8f acc[4] = {};
  v16h a = load16h(arow);
  v16h b[4];
#pragma unroll
  for (int g = 0; g < 4; ++g) b[g] = load16h(wrow + (size_t)g * HID * HID);

#pragma unroll
  for (int k0 = 0; k0 < HID; k0 += 32) {
    v16h an;
    v16h bn[4];
    if (k0 + 32 < HID) {
      an = load16h(arow + k0 + 32);
#pragma unroll
      for (int g = 0; g < 4; ++g)
        bn[g] = load16h(wrow + (size_t)g * HID * HID + k0 + 32);
    }
#pragma unroll
    for (int g = 0; g < 4; ++g)
      acc[g] = __builtin_amdgcn_wmma_f32_16x16x32_f16(false, a, false, b[g],
                                                      (short)0, acc[g], false, false);
    if (k0 + 32 < HID) {
      a = an;
#pragma unroll
      for (int g = 0; g < 4; ++g) b[g] = bn[g];
    }
  }

  const int rbase = m0 + (hg << 3);
  const int col = n0 + nn;
#pragma unroll
  for (int r = 0; r < 8; ++r) {
    int row = rbase + r;
    int rowc = row < SPOOL ? row : SPOOL - 1;   // clamp padded rows
    size_t gofs = ((size_t)t * SPOOL + rowc) * HGATE + col;
    float gi = acc[0][r] + (float)gin[gofs];
    float gf = acc[1][r] + (float)gin[gofs + 256];
    float gg = acc[2][r] + (float)gin[gofs + 512];
    float go = acc[3][r] + (float)gin[gofs + 768];
    size_t sidx = (size_t)row * HID + col;
    float cn = sigm(gf) * cst[sidx] + sigm(gi) * tanhf(gg);
    cst[sidx] = cn;
    float hn = sigm(go) * tanhf(cn);
    hout[sidx] = (_Float16)hn;
    if (row < SPOOL)
      xout[((size_t)t * SPOOL + row) * 512 + dir * HID + col] = (_Float16)hn;
  }
}

// ------------------------------------------------- fused flash attention ----
// One wave per (q-tile, batch, head). Online softmax with /sqrt(DH) and /rpe.
// K tiles for the NEXT chunk and V tiles for the CURRENT chunk are prefetched
// before the softmax VALU work so their latency hides behind it. V is consumed
// pre-transposed (vT[b][h*64+d][s], stride SPPAD) -> contiguous b128 loads.

__global__ __launch_bounds__(32) void attention_kernel(
    const _Float16* __restrict__ qb, const _Float16* __restrict__ kb,
    const _Float16* __restrict__ vT, const float* __restrict__ rpe,
    _Float16* __restrict__ cat) {
  const int lane = threadIdx.x & 31;
  const int nn = lane & 15, hg = lane >> 4;
  const int s0 = blockIdx.x << 4;
  const int b = blockIdx.y;
  const int head = blockIdx.z;
  const size_t rowstride = NHEAD * DHEAD;  // 512
  const size_t base = (size_t)b * SPOOL * rowstride + (size_t)head * DHEAD;

  int qra = s0 + nn; if (qra > SPOOL - 1) qra = SPOOL - 1;
  const _Float16* qp = qb + base + (size_t)qra * rowstride + (hg << 4);
  const v16h aq0 = load16h(qp);
  const v16h aq1 = load16h(qp + 32);

  // vT pointer for this lane's d-column (nn) within this (b, head)
  const _Float16* vTn =
      vT + ((size_t)b * 512 + (size_t)head * DHEAD + nn) * SPPAD + (hg << 4);

  v8f o[4] = {};
  float mrow[8], lrow[8];
#pragma unroll
  for (int r = 0; r < 8; ++r) { mrow[r] = -1e30f; lrow[r] = 0.f; }

  __shared__ _Float16 pls[16 * 40];  // 16 rows x 32 cols, stride 40 halves

  v16h kb0lo, kb1lo, kb0hi, kb1hi;
  {
    int klo = nn, khi = 16 + nn;
    const _Float16* plo = kb + base + (size_t)klo * rowstride + (hg << 4);
    const _Float16* phi = kb + base + (size_t)khi * rowstride + (hg << 4);
    kb0lo = load16h(plo); kb1lo = load16h(plo + 32);
    kb0hi = load16h(phi); kb1hi = load16h(phi + 32);
  }

  for (int kc = 0; kc < SPOOL; kc += 32) {
    v8f slo = {}, shi = {};
    slo = __builtin_amdgcn_wmma_f32_16x16x32_f16(false, aq0, false, kb0lo, (short)0, slo, false, false);
    slo = __builtin_amdgcn_wmma_f32_16x16x32_f16(false, aq1, false, kb1lo, (short)0, slo, false, false);
    shi = __builtin_amdgcn_wmma_f32_16x16x32_f16(false, aq0, false, kb0hi, (short)0, shi, false, false);
    shi = __builtin_amdgcn_wmma_f32_16x16x32_f16(false, aq1, false, kb1hi, (short)0, shi, false, false);

    // prefetch V tiles for this chunk (pad rows >=341 are zero-filled)
    v16h bv[4];
#pragma unroll
    for (int dt = 0; dt < 4; ++dt)
      bv[dt] = load16h(vTn + (size_t)dt * 16 * SPPAD + kc);

    // prefetch K tiles for the next chunk
    if (kc + 32 < SPOOL) {
      int klo = kc + 32 + nn, khi = kc + 48 + nn;
      if (klo > SPOOL - 1) klo = SPOOL - 1;
      if (khi > SPOOL - 1) khi = SPOOL - 1;
      const _Float16* plo = kb + base + (size_t)klo * rowstride + (hg << 4);
      const _Float16* phi = kb + base + (size_t)khi * rowstride + (hg << 4);
      kb0lo = load16h(plo); kb1lo = load16h(plo + 32);
      kb0hi = load16h(phi); kb1hi = load16h(phi + 32);
    }

    const int klo = kc + nn, khi = kc + 16 + nn;
    __syncthreads();
#pragma unroll
    for (int r = 0; r < 8; ++r) {
      int qrow = s0 + r + (hg << 3);
      int qrc = qrow > SPOOL - 1 ? SPOOL - 1 : qrow;
      float vlo = (klo < SPOOL) ? (slo[r] * 0.125f / rpe[qrc * SPOOL + klo]) : -1e30f;
      float vhi = (khi < SPOOL) ? (shi[r] * 0.125f / rpe[qrc * SPOOL + khi]) : -1e30f;
      float mx = redmax16(fmaxf(vlo, vhi));
      float mn = fmaxf(mrow[r], mx);
      float corr = __expf(mrow[r] - mn);
      mrow[r] = mn;
      float plo = __expf(vlo - mn);
      float phi = __expf(vhi - mn);
      float ps = redsum16(plo + phi);
      lrow[r] = lrow[r] * corr + ps;
      o[0][r] *= corr; o[1][r] *= corr; o[2][r] *= corr; o[3][r] *= corr;
      int lr = r + (hg << 3);
      pls[lr * 40 + nn] = (_Float16)plo;
      pls[lr * 40 + 16 + nn] = (_Float16)phi;
    }
    __syncthreads();
    v16h ap = load16h(&pls[nn * 40 + (hg << 4)]);
#pragma unroll
    for (int dt = 0; dt < 4; ++dt)
      o[dt] = __builtin_amdgcn_wmma_f32_16x16x32_f16(false, ap, false, bv[dt],
                                                     (short)0, o[dt], false, false);
  }
#pragma unroll
  for (int dt = 0; dt < 4; ++dt)
#pragma unroll
    for (int r = 0; r < 8; ++r) {
      int qrow = s0 + r + (hg << 3);
      if (qrow < SPOOL) {
        float xv = fmaxf(o[dt][r] / lrow[r], 0.f);
        cat[base + (size_t)qrow * rowstride + (dt << 4) + nn] = (_Float16)xv;
      }
    }
}

// ---------------------------------------------------------------- launch ----

extern "C" void kernel_launch(void* const* d_in, const int* in_sizes, int n_in,
                              void* d_out, int out_size, void* d_ws, size_t ws_size,
                              hipStream_t stream) {
  (void)in_sizes; (void)n_in; (void)out_size; (void)ws_size;
  const float* x_in   = (const float*)d_in[0];
  const float* conv_w = (const float*)d_in[1];
  const float* gamma  = (const float*)d_in[2];
  const float* beta   = (const float*)d_in[3];
  const float* Wih0   = (const float*)d_in[4];
  const float* Whh0   = (const float*)d_in[5];
  const float* b0     = (const float*)d_in[6];
  const float* Wih1   = (const float*)d_in[7];
  const float* Whh1   = (const float*)d_in[8];
  const float* b1     = (const float*)d_in[9];
  const float* Qw     = (const float*)d_in[10];
  const float* Qb     = (const float*)d_in[11];
  const float* Kw     = (const float*)d_in[12];
  const float* Kb     = (const float*)d_in[13];
  const float* Vw     = (const float*)d_in[14];
  const float* Vb     = (const float*)d_in[15];
  const float* MHw    = (const float*)d_in[16];
  const float* MHb    = (const float*)d_in[17];

  char* ws = (char*)d_ws;
  size_t off = 0;
  auto take = [&](size_t bytes) -> char* {
    char* p = ws + off;
    off = (off + bytes + 255) & ~(size_t)255;
    return p;
  };

  _Float16* wih0h = (_Float16*)take(2ull * 1024 * 256 * 2);
  _Float16* whh0h = (_Float16*)take(2ull * 1024 * 256 * 2);
  _Float16* wih1h = (_Float16*)take(2ull * 1024 * 512 * 2);
  _Float16* whh1h = (_Float16*)take(2ull * 1024 * 256 * 2);
  _Float16* qwh   = (_Float16*)take(512ull * 512 * 2);
  _Float16* kwh   = (_Float16*)take(512ull * 512 * 2);
  _Float16* vwh   = (_Float16*)take(512ull * 512 * 2);
  _Float16* mhwh  = (_Float16*)take(512ull * 512 * 2);
  _Float16* x1    = (_Float16*)take((size_t)MROWS * NFLT * 2);
  _Float16* gin   = (_Float16*)take(2ull * MROWS * HGATE * 2);
  _Float16* x2    = (_Float16*)take((size_t)MROWS * 512 * 2);
  _Float16* x3    = (_Float16*)take((size_t)MROWS * 512 * 2);
  _Float16* qact  = (_Float16*)take((size_t)MROWS * 512 * 2);
  _Float16* kact  = (_Float16*)take((size_t)MROWS * 512 * 2);
  _Float16* vT    = (_Float16*)take(32ull * 512 * SPPAD * 2);  // [b][h*64+d][s]
  _Float16* catb  = (_Float16*)take((size_t)MROWS * 512 * 2);
  _Float16* hA    = (_Float16*)take(2ull * SPPAD * HID * 2);
  _Float16* hB    = (_Float16*)take(2ull * SPPAD * HID * 2);
  float*    cst   = (float*)take(2ull * SPPAD * HID * 4);
  float*    rpe   = (float*)take((size_t)SPOOL * SPOOL * 4);

  auto cvt = [&](const float* s, _Float16* d, int n) {
    f32_to_f16_kernel<<<dim3((n + 255) / 256), dim3(256), 0, stream>>>(s, d, n);
  };
  cvt(Wih0, wih0h, 2 * 1024 * 256);
  cvt(Whh0, whh0h, 2 * 1024 * 256);
  cvt(Wih1, wih1h, 2 * 1024 * 512);
  cvt(Whh1, whh1h, 2 * 1024 * 256);
  cvt(Qw, qwh, 512 * 512);
  cvt(Kw, kwh, 512 * 512);
  cvt(Vw, vwh, 512 * 512);
  cvt(MHw, mhwh, 512 * 512);

  conv_bn_pool_kernel<<<dim3(MROWS), dim3(256), 0, stream>>>(x_in, conv_w, gamma, beta, x1);
  rpe_kernel<<<dim3((SPOOL * SPOOL + 255) / 256), dim3(256), 0, stream>>>(rpe);

  const int MT = MROWS / 16;  // 682

  // Layer-1 input projections (bias = merged b_ih+b_hh), both directions.
  for (int dir = 0; dir < 2; ++dir)
    gemm_wmma_kernel<1, false><<<dim3(MT, 16), dim3(32), 0, stream>>>(
        x1, wih0h + (size_t)dir * 1024 * 256, b0 + dir * 1024,
        gin + (size_t)dir * MROWS * HGATE, HGATE, 256);

  hipMemsetAsync(hA, 0, 2ull * SPPAD * HID * 2, stream);
  hipMemsetAsync(hB, 0, 2ull * SPPAD * HID * 2, stream);
  hipMemsetAsync(cst, 0, 2ull * SPPAD * HID * 4, stream);

  _Float16* hping[2] = {hA, hB};
  for (int t = 0; t < 32; ++t)
    lstm_cell_kernel<<<dim3(22, 16, 2), dim3(32), 0, stream>>>(
        gin, whh0h, hping[t & 1], hping[(t + 1) & 1], cst, x2, t);

  // Layer-2 input projections (overwrite gin).
  for (int dir = 0; dir < 2; ++dir)
    gemm_wmma_kernel<1, false><<<dim3(MT, 16), dim3(32), 0, stream>>>(
        x2, wih1h + (size_t)dir * 1024 * 512, b1 + dir * 1024,
        gin + (size_t)dir * MROWS * HGATE, HGATE, 512);

  hipMemsetAsync(hA, 0, 2ull * SPPAD * HID * 2, stream);
  hipMemsetAsync(hB, 0, 2ull * SPPAD * HID * 2, stream);
  hipMemsetAsync(cst, 0, 2ull * SPPAD * HID * 4, stream);

  for (int t = 0; t < 32; ++t)
    lstm_cell_kernel<<<dim3(22, 16, 2), dim3(32), 0, stream>>>(
        gin, whh1h, hping[t & 1], hping[(t + 1) & 1], cst, x3, t);

  // QKV projections: heads folded into N. V goes out transposed+zero-padded.
  hipMemsetAsync(vT, 0, 32ull * 512 * SPPAD * 2, stream);
  gemm_wmma_kernel<1, false><<<dim3(MT, 8), dim3(32), 0, stream>>>(x3, qwh, Qb, qact, 512, 512);
  gemm_wmma_kernel<1, false><<<dim3(MT, 8), dim3(32), 0, stream>>>(x3, kwh, Kb, kact, 512, 512);
  gemm_wmma_kernel<2, false><<<dim3(MT, 8), dim3(32), 0, stream>>>(x3, vwh, Vb, vT, 512, 512);

  attention_kernel<<<dim3(22, 32, 8), dim3(32), 0, stream>>>(qact, kact, vT, rpe, catb);

  // Output projection + ReLU, f32 straight into d_out [B,SP,512].
  gemm_wmma_kernel<0, true><<<dim3(MT, 8), dim3(32), 0, stream>>>(
      catb, mhwh, MHb, d_out, 512, 512);
}
